// MGCN_63084479644116
// MI455X (gfx1250) — matrix operand order repaired and checked by reading, not compiled, not run on hardware
//
#include <hip/hip_runtime.h>

typedef unsigned short u16;
typedef __attribute__((ext_vector_type(16))) __bf16 v16bf;
typedef __attribute__((ext_vector_type(8)))  float  v8f;

#define NN    4096
#define DD    1024
#define NOUT  1000
#define NOUTP 1024
#define KSEL  819

__device__ __forceinline__ u16 f2bf(float x) {
  unsigned int u = __float_as_uint(x);
  u += 0x7FFFu + ((u >> 16) & 1u);          // round-to-nearest-even
  return (u16)(u >> 16);
}

// ---------------- row L2 norms + bf16 pack of features ----------------
__global__ __launch_bounds__(256) void k_rownorm_bf16(const float* __restrict__ F,
                                                      u16* __restrict__ Fb,
                                                      float* __restrict__ sq) {
  int row = blockIdx.x, t = threadIdx.x;
  const float* fr = F + (size_t)row * DD;
  u16* fb = Fb + (size_t)row * DD;
  float s = 0.f;
  for (int e = 0; e < DD / 256; ++e) {
    int j = t + e * 256;
    float v = fr[j];
    s += v * v;
    fb[j] = f2bf(v);
  }
  __shared__ float red[256];
  red[t] = s; __syncthreads();
  for (int off = 128; off > 0; off >>= 1) { if (t < off) red[t] += red[t + off]; __syncthreads(); }
  if (t == 0) sq[row] = red[0];
}

// ---------------- W [D][1000] -> Wt bf16 [1024(pad N)][D] ----------------
__global__ __launch_bounds__(256) void k_wt_bf16(const float* __restrict__ W,
                                                 u16* __restrict__ Wt) {
  int idx = blockIdx.x * 256 + threadIdx.x;   // over 1024*1024
  int n = idx >> 10, k = idx & 1023;
  Wt[idx] = (n < NOUT) ? f2bf(W[(size_t)k * NOUT + n]) : (u16)0;
}

// ---------------- f32 [rows][cols] -> bf16 transposed [cols][rows] ----------------
__global__ __launch_bounds__(256) void k_convt(const float* __restrict__ in,
                                               u16* __restrict__ out,
                                               int rows, int cols) {
  size_t idx = (size_t)blockIdx.x * 256 + threadIdx.x;
  int m = (int)(idx / cols);
  int n = (int)(idx - (size_t)m * cols);
  out[(size_t)n * rows + m] = f2bf(in[idx]);
}

// ---------------- bf16 WMMA GEMM: C[M][N] = A[M][K] * Bt[N][K]^T ----------------
// MODE 0: plain f32 store. MODE 1: gaussian affinity epilogue exp(-max(|ai|^2+|aj|^2-2c,0)/9)
template <int MODE>
__global__ __launch_bounds__(256) void k_gemm_bf16(const u16* __restrict__ A,
                                                   const u16* __restrict__ Bt,
                                                   float* __restrict__ C,
                                                   int Kd, int lda, int ldb, int ldc,
                                                   const float* __restrict__ sq) {
  __shared__ u16 As[128 * 32];
  __shared__ u16 Bs[128 * 32];
  const int t = threadIdx.x;
  const int lane = t & 31, wave = t >> 5;
  const int waveM = wave >> 2, waveN = wave & 3;       // 2 x 4 wave grid
  const int m0 = blockIdx.y * 128, n0 = blockIdx.x * 128;
  const int ldRow = t >> 1, seg = t & 1;               // 2 threads load one 32-elem row
  const int half = lane >> 4, l16 = lane & 15;

  v8f acc[4][2];
  v8f zero = {0.f, 0.f, 0.f, 0.f, 0.f, 0.f, 0.f, 0.f};
  for (int mi = 0; mi < 4; ++mi)
    for (int ni = 0; ni < 2; ++ni) acc[mi][ni] = zero;

  const int ktiles = Kd >> 5;
  for (int kt = 0; kt < ktiles; ++kt) {
    const int k0 = kt << 5;
    const float4* ga = (const float4*)(A  + (size_t)(m0 + ldRow) * lda + k0 + seg * 16);
    const float4* gb = (const float4*)(Bt + (size_t)(n0 + ldRow) * ldb + k0 + seg * 16);
    float4 a0 = ga[0], a1 = ga[1];
    float4 b0 = gb[0], b1 = gb[1];
    if (kt + 1 < ktiles) {                 // global_prefetch_b8 for next K-tile
      __builtin_prefetch((const void*)(A  + (size_t)(m0 + ldRow) * lda + k0 + 32 + seg * 16), 0, 1);
      __builtin_prefetch((const void*)(Bt + (size_t)(n0 + ldRow) * ldb + k0 + 32 + seg * 16), 0, 1);
    }
    float4* sa = (float4*)&As[ldRow * 32 + seg * 16]; sa[0] = a0; sa[1] = a1;
    float4* sb = (float4*)&Bs[ldRow * 32 + seg * 16]; sb[0] = b0; sb[1] = b1;
    __syncthreads();

    union FU { float4 f[2]; v16bf v; };
    v16bf afr[4], bfr[2];
    // A fragment (16x32): lane half selects K runs {half*8..+7, 16+half*8..+7}
    for (int mi = 0; mi < 4; ++mi) {
      int rA = waveM * 64 + mi * 16 + l16;
      FU u;
      u.f[0] = *(const float4*)&As[rA * 32 + half * 8];
      u.f[1] = *(const float4*)&As[rA * 32 + half * 8 + 16];
      afr[mi] = u.v;
    }
    // B fragment (32x16): lane half selects K runs {half*8..+7, 16+half*8..+7}
    for (int ni = 0; ni < 2; ++ni) {
      int rB = waveN * 32 + ni * 16 + l16;
      FU u;
      u.f[0] = *(const float4*)&Bs[rB * 32 + half * 8];
      u.f[1] = *(const float4*)&Bs[rB * 32 + half * 8 + 16];
      bfr[ni] = u.v;
    }
#pragma unroll
    for (int mi = 0; mi < 4; ++mi)
#pragma unroll
      for (int ni = 0; ni < 2; ++ni)
        acc[mi][ni] = __builtin_amdgcn_wmma_f32_16x16x32_bf16(
            false, afr[mi], false, bfr[ni], (short)0, acc[mi][ni], false, false);
    __syncthreads();
  }

  // store: D element r of VGPR block -> m = base + (lane>=16?8:0) + r, n = base + lane%16
  for (int mi = 0; mi < 4; ++mi)
    for (int ni = 0; ni < 2; ++ni) {
      int n = n0 + waveN * 32 + ni * 16 + l16;
      int mbase = m0 + waveM * 64 + mi * 16 + half * 8;
#pragma unroll
      for (int r = 0; r < 8; ++r) {
        int m = mbase + r;
        float c = acc[mi][ni][r];
        if (MODE == 1) {
          float d2 = fmaxf(sq[m] + sq[n] - 2.f * c, 0.f);
          C[(size_t)m * ldc + n] = __expf(-d2 * (1.f / 9.f));
        } else {
          C[(size_t)m * ldc + n] = c;
        }
      }
    }
}

// ---------------- per-row K-th-largest threshold via bisection ----------------
__global__ __launch_bounds__(256) void k_topk(const float* __restrict__ Ae,
                                              float* __restrict__ thr) {
  __shared__ float rowv[NN];
  __shared__ int cnt;
  __shared__ float s_lo, s_hi;
  int i = blockIdx.x, t = threadIdx.x;
  const float* ar = Ae + (size_t)i * NN;
  for (int e = 0; e < NN / 256; ++e) rowv[t + e * 256] = ar[t + e * 256];
  if (t == 0) { s_lo = 0.f; s_hi = 1.0000001f; }
  __syncthreads();
  for (int it = 0; it < 25; ++it) {
    if (t == 0) cnt = 0;
    __syncthreads();
    float mid = 0.5f * (s_lo + s_hi);
    int lc = 0;
    for (int e = 0; e < NN / 256; ++e) lc += (rowv[t + e * 256] >= mid) ? 1 : 0;
    atomicAdd(&cnt, lc);
    __syncthreads();
    if (t == 0) { if (cnt >= KSEL) s_lo = mid; else s_hi = mid; }
    __syncthreads();
  }
  if (t == 0) thr[i] = s_lo;
}

// ---------------- mutual-kNN mask + eye, in place; rowsum -> 1/d ----------------
__global__ __launch_bounds__(256) void k_buildA(float* __restrict__ Ae,
                                                const float* __restrict__ thr,
                                                float* __restrict__ rinv) {
  int i = blockIdx.x, t = threadIdx.x;
  float ti = thr[i];
  float* ar = Ae + (size_t)i * NN;
  float s = 0.f;
  for (int e = 0; e < NN / 256; ++e) {
    int j = t + e * 256;
    float a = ar[j];
    float v = (j == i) ? 1.f : ((a >= ti && a >= thr[j]) ? a : 0.f);
    ar[j] = v;
    s += v;
  }
  __shared__ float red[256];
  red[t] = s; __syncthreads();
  for (int off = 128; off > 0; off >>= 1) { if (t < off) red[t] += red[t + off]; __syncthreads(); }
  if (t == 0) rinv[i] = rsqrtf(red[0] + 1.f);
}

// ---------------- An = A/(d_i d_j) packed to bf16 ----------------
__global__ __launch_bounds__(256) void k_anb(const float* __restrict__ A,
                                             const float* __restrict__ rinv,
                                             u16* __restrict__ Anb) {
  size_t idx = (size_t)blockIdx.x * 256 + threadIdx.x;
  int i = (int)(idx >> 12);
  int j = (int)(idx & 4095);
  Anb[idx] = f2bf(A[idx] * rinv[i] * rinv[j]);
}

// ---------------- aifa mix + bias + BN(eval) + ReLU ----------------
__global__ __launch_bounds__(256) void k_epilogue(const float* __restrict__ sup,
                                                  const float* __restrict__ y1,
                                                  const float* __restrict__ y2,
                                                  const float* __restrict__ bias,
                                                  const float* __restrict__ a1p,
                                                  const float* __restrict__ a2p,
                                                  const float* __restrict__ a3p,
                                                  const float* __restrict__ gamma,
                                                  const float* __restrict__ beta,
                                                  const float* __restrict__ mean,
                                                  const float* __restrict__ var,
                                                  float* __restrict__ out) {
  int idx = blockIdx.x * 256 + threadIdx.x;
  if (idx >= NN * NOUT) return;
  int m = idx / NOUT, c = idx - m * NOUT;
  float e1 = a1p[0], e2 = a2p[0], e3 = a3p[0];
  float mx = fmaxf(e1, fmaxf(e2, e3));
  float x1 = __expf(e1 - mx), x2 = __expf(e2 - mx), x3 = __expf(e3 - mx);
  float inv = 1.f / (x1 + x2 + x3);
  size_t o = (size_t)m * NOUTP + c;
  float x = x1 * inv * sup[o] + x2 * inv * y1[o] + x3 * inv * y2[o] + bias[c];
  x = (x - mean[c]) * rsqrtf(var[c] + 1e-5f) * gamma[c] + beta[c];
  out[idx] = fmaxf(x, 0.f);
}

extern "C" void kernel_launch(void* const* d_in, const int* in_sizes, int n_in,
                              void* d_out, int out_size, void* d_ws, size_t ws_size,
                              hipStream_t stream) {
  const float* F     = (const float*)d_in[0];
  const float* W     = (const float*)d_in[1];
  const float* bias  = (const float*)d_in[2];
  const float* a1    = (const float*)d_in[3];
  const float* a2    = (const float*)d_in[4];
  const float* a3    = (const float*)d_in[5];
  const float* gamma = (const float*)d_in[6];
  const float* beta  = (const float*)d_in[7];
  const float* mean  = (const float*)d_in[8];
  const float* var   = (const float*)d_in[9];
  float* out = (float*)d_out;
  char* ws = (char*)d_ws;

  // workspace layout (first 64 MiB holds Ae/A, later reused for support/y1/y2)
  float* Ae   = (float*)(ws + 0);
  float* sup  = (float*)(ws + 0);
  float* y1   = (float*)(ws + ((size_t)16 << 20));
  float* y2   = (float*)(ws + ((size_t)32 << 20));
  u16*   Anb  = (u16*)  (ws + ((size_t)64 << 20));
  u16*   Fb   = (u16*)  (ws + ((size_t)96 << 20));
  u16*   Wt   = (u16*)  (ws + ((size_t)104 << 20));
  u16*   SbT  = (u16*)  (ws + ((size_t)106 << 20));
  u16*   Y1bT = (u16*)  (ws + ((size_t)114 << 20));
  float* sq   = (float*)(ws + ((size_t)122 << 20));
  float* thr  = (float*)(ws + ((size_t)122 << 20) + 16384);
  float* rinv = (float*)(ws + ((size_t)122 << 20) + 32768);

  // 1) pack features to bf16 + row norms
  k_rownorm_bf16<<<NN, 256, 0, stream>>>(F, Fb, sq);
  // 2) pack W transposed/padded to bf16
  k_wt_bf16<<<(DD * NOUTP) / 256, 256, 0, stream>>>(W, Wt);
  // 3) Gram GEMM (bf16 WMMA) with gaussian-affinity epilogue -> Ae
  k_gemm_bf16<1><<<dim3(NN / 128, NN / 128), 256, 0, stream>>>(Fb, Fb, Ae, DD, DD, DD, NN, sq);
  // 4) per-row top-K threshold
  k_topk<<<NN, 256, 0, stream>>>(Ae, thr);
  // 5) mutual-kNN + eye, rowsum -> 1/d
  k_buildA<<<NN, 256, 0, stream>>>(Ae, thr, rinv);
  // 6) normalized adjacency to bf16
  k_anb<<<((size_t)NN * NN) / 256, 256, 0, stream>>>(Ae, rinv, Anb);
  // 7) support = F * W  (bf16 WMMA)
  k_gemm_bf16<0><<<dim3(NOUTP / 128, NN / 128), 256, 0, stream>>>(Fb, Wt, sup, DD, DD, DD, NOUTP, nullptr);
  k_convt<<<(NN * NOUTP) / 256, 256, 0, stream>>>(sup, SbT, NN, NOUTP);
  // 8) y1 = An * support
  k_gemm_bf16<0><<<dim3(NOUTP / 128, NN / 128), 256, 0, stream>>>(Anb, SbT, y1, NN, NN, NN, NOUTP, nullptr);
  k_convt<<<(NN * NOUTP) / 256, 256, 0, stream>>>(y1, Y1bT, NN, NOUTP);
  // 9) y2 = An * y1
  k_gemm_bf16<0><<<dim3(NOUTP / 128, NN / 128), 256, 0, stream>>>(Anb, Y1bT, y2, NN, NN, NN, NOUTP, nullptr);
  // 10) aifa mix + bias + BN + ReLU
  k_epilogue<<<(NN * NOUT + 255) / 256, 256, 0, stream>>>(sup, y1, y2, bias, a1, a2, a3,
                                                          gamma, beta, mean, var, out);
}
